// MultiHeadAttention_34565896798411
// MI455X (gfx1250) — matrix operand order repaired and checked by reading, not compiled, not run on hardware
//
#include <hip/hip_runtime.h>
#include <hip/hip_bf16.h>
#include <math.h>

// ---------------------------------------------------------------------------
// Fused attention block for MI455X (gfx1250, wave32, WMMA 16x16x32 bf16).
// Round 3: flash attention restructured so a workgroup (8 waves) covers 128
// consecutive queries of one (b,h); each 16-key K/V tile is staged ONCE into
// LDS with GLOBAL_LOAD_ASYNC_TO_LDS_B128 (double buffered, s_wait_asynccnt +
// workgroup barrier), cutting K/V global traffic 8x and exercising the CDNA5
// async-to-LDS path.  GEMMs keep the round-2 form: bf16 pre-packed operands,
// all-b128 fetches, 16x64 register blocking, global_prefetch streaming.
//
// Pipeline:
//   P0) pack: x -> bf16 ; Wqkv -> bf16 transposed [N,K]; Wout -> bf16 [N,K]
//   1)  qkv GEMM -> Qf32/Kf32 [B,H,S,HD], Vt bf16 [B,H,HD,S]
//   2)  RoPE(Q,K) -> Qb bf16 (pre-scaled 1/sqrt(HD)), Kb bf16 [B,H,S,HD]
//   3)  flash attention (+-128 window) with async-LDS K/V tiles -> Ab bf16
//   4)  out GEMM + bias -> d_out f32
// Workspace: ~61.4 MB of d_ws.
// ---------------------------------------------------------------------------

typedef __attribute__((ext_vector_type(16))) __bf16 v16bf;
typedef __attribute__((ext_vector_type(8)))  __bf16 v8bf;
typedef __attribute__((ext_vector_type(8)))  float  v8f;

constexpr int Bc  = 2;
constexpr int Sc  = 2048;
constexpr int Dc  = 768;
constexpr int Hc  = 12;
constexpr int HDc = 64;
constexpr int ROTc = 64;
constexpr int WIN  = 128;
constexpr int GEVERY = 3;
constexpr float GTHETA = 160000.0f;
constexpr float LTHETA = 10000.0f;

__device__ inline v16bf cat16(v8bf lo, v8bf hi) {
  return __builtin_shufflevector(lo, hi, 0, 1, 2, 3, 4, 5, 6, 7,
                                 8, 9, 10, 11, 12, 13, 14, 15);
}
__device__ inline v8bf zero8() {
  v8bf z;
#pragma unroll
  for (int i = 0; i < 8; ++i) z[i] = (__bf16)0.0f;
  return z;
}
__device__ inline v16bf ld16(const __bf16* __restrict__ p) {
  return cat16(*(const v8bf*)p, *(const v8bf*)(p + 8));
}

// --- CDNA5 async copy: global -> LDS, 16 bytes per lane (ASYNCcnt). --------
__device__ inline void async_copy_b128(unsigned lds_addr, const __bf16* gaddr) {
  asm volatile("global_load_async_to_lds_b128 %0, %1, off"
               :
               : "v"(lds_addr), "v"(gaddr)
               : "memory");
}
__device__ inline void wait_asynccnt0() {
  asm volatile("s_wait_asynccnt 0x0" ::: "memory");
}
__device__ inline unsigned lds_addr_of(const void* p) {
  return (unsigned)(uintptr_t)p;   // generic LDS pointer: low 32b = LDS offset
}

// ---------------------------------------------------------------------------
// P0a: elementwise f32 -> bf16 (same layout).
// ---------------------------------------------------------------------------
__global__ __launch_bounds__(256) void cvt_bf16_kernel(
    const float* __restrict__ src, __bf16* __restrict__ dst, int n) {
  const int i = blockIdx.x * blockDim.x + threadIdx.x;
  if (i < n) dst[i] = (__bf16)src[i];
}

// ---------------------------------------------------------------------------
// P0b: W[K,N] f32 -> Wt[N,K] bf16 (so B-operand K-runs are contiguous).
// ---------------------------------------------------------------------------
__global__ __launch_bounds__(256) void transpose_bf16_kernel(
    const float* __restrict__ src, __bf16* __restrict__ dst, int K, int N) {
  const int i = blockIdx.x * blockDim.x + threadIdx.x;
  if (i >= K * N) return;
  const int n = i % N, k = i / N;
  dst[(size_t)n * K + k] = (__bf16)src[i];
}

// ---------------------------------------------------------------------------
// Core GEMM micro-tile: 16x64 per wave (4 x 16x16 WMMA sharing one A fetch).
// ---------------------------------------------------------------------------
__device__ inline void wmma_gemm_16x64(const __bf16* __restrict__ A, int lda,
                                       const __bf16* __restrict__ Wt, int ldk,
                                       int mBase, int nBase, int K, int lane,
                                       v8f acc[4]) {
  const int col = lane & 15;
  const int hi  = lane >> 4;
  const __bf16* arow = A + (size_t)(mBase + col) * lda;
  const __bf16* wp0 = Wt + (size_t)(nBase + 0 * 16 + col) * ldk + 16 * hi;
  const __bf16* wp1 = Wt + (size_t)(nBase + 1 * 16 + col) * ldk + 16 * hi;
  const __bf16* wp2 = Wt + (size_t)(nBase + 2 * 16 + col) * ldk + 16 * hi;
  const __bf16* wp3 = Wt + (size_t)(nBase + 3 * 16 + col) * ldk + 16 * hi;
  for (int k0 = 0; k0 < K; k0 += 32) {
    if (k0 + 256 < K) {
      __builtin_prefetch(arow + k0 + 256, 0, 1);
      __builtin_prefetch(wp0 + k0 + 256, 0, 1);
    }
    const v16bf a = cat16(*(const v8bf*)(arow + k0 + 16 * hi),
                          *(const v8bf*)(arow + k0 + 16 * hi + 8));
    acc[0] = __builtin_amdgcn_wmma_f32_16x16x32_bf16(false, a, false, ld16(wp0 + k0),
                                                     (short)0, acc[0], false, false);
    acc[1] = __builtin_amdgcn_wmma_f32_16x16x32_bf16(false, a, false, ld16(wp1 + k0),
                                                     (short)0, acc[1], false, false);
    acc[2] = __builtin_amdgcn_wmma_f32_16x16x32_bf16(false, a, false, ld16(wp2 + k0),
                                                     (short)0, acc[2], false, false);
    acc[3] = __builtin_amdgcn_wmma_f32_16x16x32_bf16(false, a, false, ld16(wp3 + k0),
                                                     (short)0, acc[3], false, false);
  }
}

// ---------------------------------------------------------------------------
// Kernel 1: QKV projection, one wave per 16x64 tile of [B*S, 3D].
// ---------------------------------------------------------------------------
__global__ __launch_bounds__(256) void qkv_proj_kernel(
    const __bf16* __restrict__ xb, const __bf16* __restrict__ WqkvT,
    const float* __restrict__ bqkv, float* __restrict__ Qf,
    float* __restrict__ Kf, __bf16* __restrict__ Vt) {
  const int wave = (blockIdx.x * blockDim.x + threadIdx.x) >> 5;
  const int lane = threadIdx.x & 31;
  const int nGroups = (3 * Dc) / 64;               // 36
  const int mt = wave / nGroups, ng = wave % nGroups;
  const int mBase = mt * 16, nBase = ng * 64;

  v8f acc[4] = {{}, {}, {}, {}};
  wmma_gemm_16x64(xb, Dc, WqkvT, Dc, mBase, nBase, Dc, lane, acc);

  const int col = lane & 15, hi = lane >> 4;
#pragma unroll
  for (int t = 0; t < 4; ++t) {
    const int gcol = nBase + t * 16 + col;
    const float bias = bqkv[gcol];
    const int sel = gcol / Dc;
    const int d = gcol % Dc;
    const int h = d / HDc, hd = d % HDc;
#pragma unroll
    for (int r = 0; r < 8; ++r) {
      const int m = mBase + r + 8 * hi;
      const int bb = m / Sc, ss = m % Sc;
      const float v = acc[t][r] + bias;
      if (sel == 0)
        Qf[(((size_t)bb * Hc + h) * Sc + ss) * HDc + hd] = v;
      else if (sel == 1)
        Kf[(((size_t)bb * Hc + h) * Sc + ss) * HDc + hd] = v;
      else
        Vt[(((size_t)bb * Hc + h) * HDc + hd) * Sc + ss] = (__bf16)v;
    }
  }
}

// ---------------------------------------------------------------------------
// Kernel 2: RoPE on all 64 dims; writes bf16 Q (pre-scaled 1/sqrt(HD)) and K.
// ---------------------------------------------------------------------------
__global__ __launch_bounds__(256) void rope_pack_kernel(
    const float* __restrict__ Qf, const float* __restrict__ Kf,
    __bf16* __restrict__ Qb, __bf16* __restrict__ Kb,
    const int* __restrict__ layer_idx) {
  const int idx = blockIdx.x * blockDim.x + threadIdx.x;
  const int total = Bc * Hc * Sc * (ROTc / 2);
  if (idx >= total) return;
  const int i = idx % (ROTc / 2);
  int rest = idx / (ROTc / 2);
  const int s = rest % Sc; rest /= Sc;
  const int h = rest % Hc;
  const int b = rest / Hc;

  const bool isg = (layer_idx[0] % GEVERY) == 0;
  const float theta = isg ? GTHETA : LTHETA;
  const float freq = __expf(-((2.0f * (float)i) / (float)ROTc) * __logf(theta));
  const float ang = (float)s * freq;
  float sn, cs;
  __sincosf(ang, &sn, &cs);

  const float qscale = 0.125f;
  const size_t base = (((size_t)b * Hc + h) * Sc + s) * HDc + 2 * i;
  const float qe = Qf[base], qo = Qf[base + 1];
  Qb[base]     = (__bf16)((qe * cs - qo * sn) * qscale);
  Qb[base + 1] = (__bf16)((qo * cs + qe * sn) * qscale);
  const float ke = Kf[base], ko = Kf[base + 1];
  Kb[base]     = (__bf16)(ke * cs - ko * sn);
  Kb[base + 1] = (__bf16)(ko * cs + ke * sn);
}

// ---------------------------------------------------------------------------
// Kernel 3: flash attention.  Workgroup = 8 waves = 128 consecutive queries
// of one (b,h).  K/V 16-key tiles staged once per workgroup into LDS via
// async copy (double buffered): 256 lanes x 16B = 2KB K-tile + 2KB V-tile.
//
// Per-wave math (unchanged): S^T = K_tile @ Q^T so the post-softmax tile is
// already in the A-operand layout for P@V (keys padded 16->32 with zeros).
// ---------------------------------------------------------------------------
__global__ __launch_bounds__(256) void flash_attn_kernel(
    const __bf16* __restrict__ Qb, const __bf16* __restrict__ Kb,
    const __bf16* __restrict__ Vt, __bf16* __restrict__ Ab,
    const int* __restrict__ layer_idx) {
  __shared__ __attribute__((aligned(16))) __bf16 ldsK[2][16 * HDc];  // [key][hd]
  __shared__ __attribute__((aligned(16))) __bf16 ldsV[2][HDc * 16];  // [hd][key]

  const int tid = threadIdx.x;
  const int wv  = tid >> 5;
  const int lane = tid & 31;
  const int col = lane & 15, hi = lane >> 4;

  const int qbCount = Sc / 128;                    // 16 query-blocks per (b,h)
  const int qb = blockIdx.x % qbCount;
  const int bh = blockIdx.x / qbCount;
  const int h = bh % Hc, b = bh / Hc;
  const bool local_mask = (layer_idx[0] % GEVERY) != 0;

  const int blkQ0 = qb * 128;
  const int q0 = blkQ0 + wv * 16;                  // this wave's query tile

  const __bf16* Qp = Qb + ((size_t)b * Hc + h) * Sc * HDc;
  const __bf16* Kp = Kb + ((size_t)b * Hc + h) * Sc * HDc;
  const __bf16* Vp = Vt + ((size_t)b * Hc + h) * HDc * Sc;   // [HD, S]

  // Q^T B-operand for this wave (global, contiguous 16-bf16 runs).
  const __bf16* qr = Qp + (size_t)(q0 + col) * HDc;
  const v16bf qb0 = ld16(qr + 16 * hi);
  const v16bf qb1 = ld16(qr + 32 + 16 * hi);

  // Key-tile range for the whole workgroup (union of the 8 waves' bands).
  int ktBegin = 0, ktEnd = Sc / 16 - 1;
  if (local_mask) {
    ktBegin = max(0, (blkQ0 - WIN) >> 4);
    ktEnd   = min(Sc / 16 - 1, (blkQ0 + 127 + WIN) >> 4);
  }

  // Cooperative async stage of tile kt into LDS buffer `buf`:
  //   tid<128 : K tile, 16 rows x 64 bf16 (128B/row) -> row-major in LDS
  //   tid>=128: V tile, 64 hd-rows x 16 keys (32B/row) -> hd-major in LDS
  auto stage = [&](int buf, int k0) {
    if (tid < 128) {
      const int row = tid >> 3, ch = (tid & 7) * 8;
      async_copy_b128(lds_addr_of(&ldsK[buf][row * HDc + ch]),
                      Kp + (size_t)(k0 + row) * HDc + ch);
    } else {
      const int tt = tid - 128;
      const int hd = tt >> 1, ch = (tt & 1) * 8;
      async_copy_b128(lds_addr_of(&ldsV[buf][hd * 16 + ch]),
                      Vp + (size_t)hd * Sc + k0 + ch);
    }
  };

  stage(0, ktBegin * 16);
  wait_asynccnt0();
  __syncthreads();

  float m_cur = -1e30f, l_cur = 0.0f;
  v8f o[4] = {{}, {}, {}, {}};

  for (int kt = ktBegin; kt <= ktEnd; ++kt) {
    const int bufIdx = (kt - ktBegin) & 1;
    if (kt < ktEnd) stage(bufIdx ^ 1, (kt + 1) * 16);   // prefetch next tile
    const int k0 = kt * 16;

    // Wave-uniform band check keeps EXEC all-ones around the WMMAs.
    const bool active =
        !local_mask || !(k0 > q0 + 15 + WIN || k0 + 15 < q0 - WIN);
    if (active) {
      // K A-operand from LDS (ds_load_b128 runs).
      const __bf16* kr = &ldsK[bufIdx][col * HDc];
      const v16bf ka0 = cat16(*(const v8bf*)(kr + 16 * hi),
                              *(const v8bf*)(kr + 16 * hi + 8));
      const v16bf ka1 = cat16(*(const v8bf*)(kr + 32 + 16 * hi),
                              *(const v8bf*)(kr + 40 + 16 * hi));
      v8f st = {};
      st = __builtin_amdgcn_wmma_f32_16x16x32_bf16(false, ka0, false, qb0,
                                                   (short)0, st, false, false);
      st = __builtin_amdgcn_wmma_f32_16x16x32_bf16(false, ka1, false, qb1,
                                                   (short)0, st, false, false);

      const int q = q0 + col;
      float sv[8];
      float tmax = -1e30f;
#pragma unroll
      for (int r = 0; r < 8; ++r) {
        float s = st[r];
        const int key = k0 + r + 8 * hi;
        if (local_mask && (key > q + WIN || key < q - WIN)) s = -1e30f;
        sv[r] = s;
        tmax = fmaxf(tmax, s);
      }
      tmax = fmaxf(tmax, __shfl_xor(tmax, 16));
      const float m_new = fmaxf(m_cur, tmax);
      const float corr = __expf(m_cur - m_new);

      v16bf pa;
      float rowsum = 0.0f;
#pragma unroll
      for (int r = 0; r < 8; ++r) {
        float p = __expf(sv[r] - m_new);
        if (sv[r] < -1e29f) p = 0.0f;
        pa[r] = (__bf16)p;
        pa[8 + r] = (__bf16)0.0f;
        rowsum += p;
      }
      rowsum += __shfl_xor(rowsum, 16);
      l_cur = l_cur * corr + rowsum;
      m_cur = m_new;

      float fr[8];
#pragma unroll
      for (int r = 0; r < 8; ++r) fr[r] = __shfl(corr, r + 8 * hi);
#pragma unroll
      for (int nt = 0; nt < 4; ++nt)
#pragma unroll
        for (int r = 0; r < 8; ++r) o[nt][r] *= fr[r];

      // P @ V from LDS: lane col = hd, 16 contiguous keys per row.
#pragma unroll
      for (int nt = 0; nt < 4; ++nt) {
        v8bf vlo, vhi;
        {
          const __bf16* vp = &ldsV[bufIdx][(nt * 16 + col) * 16];
          vlo = *(const v8bf*)(vp + 8 * hi);
          vhi = zero8();
        }
        o[nt] = __builtin_amdgcn_wmma_f32_16x16x32_bf16(
            false, pa, false, cat16(vlo, vhi), (short)0, o[nt], false, false);
      }
    }

    wait_asynccnt0();       // next tile landed in the alternate buffer
    __syncthreads();        // all waves done reading before it is overwritten
  }

  float linv[8];
#pragma unroll
  for (int r = 0; r < 8; ++r) linv[r] = 1.0f / __shfl(l_cur, r + 8 * hi);
#pragma unroll
  for (int nt = 0; nt < 4; ++nt)
#pragma unroll
    for (int r = 0; r < 8; ++r) {
      const int s = q0 + r + 8 * hi;
      const int d = h * HDc + nt * 16 + col;
      Ab[((size_t)b * Sc + s) * Dc + d] = (__bf16)(o[nt][r] * linv[r]);
    }
}

// ---------------------------------------------------------------------------
// Kernel 4: output projection, one wave per 16x64 tile of [B*S, D], f32 out.
// ---------------------------------------------------------------------------
__global__ __launch_bounds__(256) void out_proj_kernel(
    const __bf16* __restrict__ Ab, const __bf16* __restrict__ WoutT,
    const float* __restrict__ bout, float* __restrict__ C) {
  const int wave = (blockIdx.x * blockDim.x + threadIdx.x) >> 5;
  const int lane = threadIdx.x & 31;
  const int nGroups = Dc / 64;                     // 12
  const int mt = wave / nGroups, ng = wave % nGroups;
  const int mBase = mt * 16, nBase = ng * 64;

  v8f acc[4] = {{}, {}, {}, {}};
  wmma_gemm_16x64(Ab, Dc, WoutT, Dc, mBase, nBase, Dc, lane, acc);

  const int col = lane & 15, hi = lane >> 4;
#pragma unroll
  for (int t = 0; t < 4; ++t) {
    const float bias = bout[nBase + t * 16 + col];
#pragma unroll
    for (int r = 0; r < 8; ++r)
      C[(size_t)(mBase + r + 8 * hi) * Dc + nBase + t * 16 + col] =
          acc[t][r] + bias;
  }
}

// ---------------------------------------------------------------------------
extern "C" void kernel_launch(void* const* d_in, const int* in_sizes, int n_in,
                              void* d_out, int out_size, void* d_ws, size_t ws_size,
                              hipStream_t stream) {
  (void)in_sizes; (void)n_in; (void)out_size; (void)ws_size;
  const float* x    = (const float*)d_in[0];
  const float* Wqkv = (const float*)d_in[1];
  const float* bqkv = (const float*)d_in[2];
  const float* Wout = (const float*)d_in[3];
  const float* bout = (const float*)d_in[4];
  const int* layer_idx = (const int*)d_in[5];
  float* out = (float*)d_out;

  char* ws = (char*)d_ws;
  const size_t n1 = (size_t)Bc * Hc * Sc * HDc;    // 3,145,728 == B*S*D
  float*  Qf    = (float*)ws;              ws += n1 * 4;
  float*  Kf    = (float*)ws;              ws += n1 * 4;
  __bf16* Vt    = (__bf16*)ws;             ws += n1 * 2;
  __bf16* Qbuf  = (__bf16*)ws;             ws += n1 * 2;
  __bf16* Kbuf  = (__bf16*)ws;             ws += n1 * 2;
  __bf16* Ab    = (__bf16*)ws;             ws += n1 * 2;
  __bf16* xb    = (__bf16*)ws;             ws += n1 * 2;
  __bf16* WqkvT = (__bf16*)ws;             ws += (size_t)(3 * Dc) * Dc * 2;
  __bf16* WoutT = (__bf16*)ws;

  {
    const int n = Bc * Sc * Dc;
    cvt_bf16_kernel<<<(n + 255) / 256, 256, 0, stream>>>(x, xb, n);
  }
  {
    const int n = Dc * 3 * Dc;
    transpose_bf16_kernel<<<(n + 255) / 256, 256, 0, stream>>>(Wqkv, WqkvT, Dc, 3 * Dc);
  }
  {
    const int n = Dc * Dc;
    transpose_bf16_kernel<<<(n + 255) / 256, 256, 0, stream>>>(Wout, WoutT, Dc, Dc);
  }
  {
    const int waves = (Bc * Sc / 16) * ((3 * Dc) / 64);  // 9216
    qkv_proj_kernel<<<waves / 8, 256, 0, stream>>>(xb, WqkvT, bqkv, Qf, Kf, Vt);
  }
  {
    const int total = Bc * Hc * Sc * (ROTc / 2);
    rope_pack_kernel<<<(total + 255) / 256, 256, 0, stream>>>(Qf, Kf, Qbuf, Kbuf, layer_idx);
  }
  {
    const int blocks = Bc * Hc * (Sc / 128);             // 384 workgroups
    flash_attn_kernel<<<blocks, 256, 0, stream>>>(Qbuf, Kbuf, Vt, Ab, layer_idx);
  }
  {
    const int waves = (Bc * Sc / 16) * (Dc / 64);        // 3072
    out_proj_kernel<<<waves / 8, 256, 0, stream>>>(Ab, WoutT, bout, out);
  }
}
